// OT_Loss_18159121728070
// MI455X (gfx1250) — matrix operand order repaired and checked by reading, not compiled, not run on hardware
//
#include <hip/hip_runtime.h>

// ---------------------------------------------------------------------------
// Sinkhorn OT loss, MI455X (gfx1250) version.
// One 1024-thread workgroup per image; everything LDS-resident; GEMMs use
// v_wmma_f32_16x16x32_bf16 (wave32 WMMA, f32 accumulate).
// ---------------------------------------------------------------------------

typedef __bf16 bf16_t;
typedef __attribute__((ext_vector_type(16))) __bf16 v16bf;
typedef __attribute__((ext_vector_type(8)))  float  v8f;

#define WW    128
#define HH    128
#define NPTS  512
#define HWSZ  (HH * WW)
#define NITER 100
#define MEPS  1e-16f
#define TPB   1024

// LDS layout (bytes):
//   Ex  [NPTS][WW]  bf16 : 131072
//   EyT [HH][NPTS]  bf16 : 131072
//   Vb  [HH][WW]    bf16 :  32768
//   u,r,Q,px,py     f32  : 5*2048
//   scal[8]         f32  :     32
#define OFF_EX   0
#define OFF_EYT  131072
#define OFF_VB   262144
#define OFF_F32  294912
#define SMEM_BYTES (294912 + 5 * 2048 + 32)   // 305184 < 320 KB WGP LDS

__device__ __forceinline__ float xcoord(int w) { return (w + 0.5f) * (2.0f / WW) - 1.0f; }
__device__ __forceinline__ float ycoord(int h) { return (h + 0.5f) * (2.0f / HH) - 1.0f; }

__global__ __launch_bounds__(TPB)
void ot_sinkhorn_kernel(const float* __restrict__ nd_g,
                        const float* __restrict__ ud_g,
                        const float* __restrict__ pts_g,
                        float* __restrict__ ws) {
  extern __shared__ char smem[];
  bf16_t* Ex   = (bf16_t*)(smem + OFF_EX);    // Ex[n*WW + w]  = exp(-(x_n-xc_w)^2)
  bf16_t* EyT  = (bf16_t*)(smem + OFF_EYT);   // EyT[h*NPTS+n] = exp(-(y_n-yc_h)^2)
  bf16_t* Vb   = (bf16_t*)(smem + OFF_VB);    // Vb[h*WW + w]  = v (bf16)
  float*  u    = (float*)(smem + OFF_F32);
  float*  r    = u  + NPTS;
  float*  Q    = r  + NPTS;
  float*  px   = Q  + NPTS;
  float*  py   = px + NPTS;
  float*  scal = py + NPTS;                   // [0]=sc [1]=ot [2]=sb [3]=wd

  const int tid  = threadIdx.x;
  const int lane = tid & 31;
  const int wave = tid >> 5;
  const int hi   = (lane >> 4) & 1;           // which 16-lane half
  const int lcol = lane & 15;
  const int img  = blockIdx.x;

  const float* nd  = nd_g  + img * HWSZ;
  const float* ud  = ud_g  + img * HWSZ;
  const float* pts = pts_g + img * NPTS * 2;

  // ---------------- init ----------------
  if (tid < NPTS) {
    px[tid] = pts[2 * tid + 0] * (2.0f / WW) - 1.0f;
    py[tid] = pts[2 * tid + 1] * (2.0f / HH) - 1.0f;
    u[tid]  = 1.0f / NPTS;
    Q[tid]  = 0.0f;
  }
  if (tid < 8) scal[tid] = 0.0f;
  __syncthreads();

  for (int idx = tid; idx < NPTS * WW; idx += TPB) {
    int n = idx >> 7, w = idx & (WW - 1);
    float d = px[n] - xcoord(w);
    Ex[idx] = (bf16_t)__expf(-d * d);
  }
  for (int idx = tid; idx < HH * NPTS; idx += TPB) {
    int h = idx >> 9, n = idx & (NPTS - 1);
    float d = py[n] - ycoord(h);
    EyT[idx] = (bf16_t)__expf(-d * d);
  }
  {   // sc = sum(ud)
    float loc = 0.0f;
    for (int idx = tid; idx < HWSZ; idx += TPB) loc += ud[idx];
    atomicAdd(&scal[0], loc);
  }
  __syncthreads();

  // ---------------- Sinkhorn loop ----------------
#pragma unroll 1
  for (int it = 0; it < NITER; ++it) {
    const bool last = (it == NITER - 1);

    // ===== phase 1: S = (u .* Ey)^T @ Ex ; v = nd / (S + eps) =====
    float ot_loc = 0.0f, sb_loc = 0.0f;
    for (int t = wave * 2; t < wave * 2 + 2; ++t) {   // 64 tiles / 32 waves
      const int h0 = (t >> 3) * 16;
      const int w0 = (t & 7) * 16;
      v8f c = {};
      for (int kb = 0; kb < NPTS; kb += 32) {
        v16bf a, b;
#pragma unroll
        for (int jj = 0; jj < 8; ++jj) {
          const int kloc = ((jj < 4) ? 0 : 16) + hi * 8 + 2 * (jj & 3);
          const int k = kb + kloc;
          // A[m=h][k=n] = u[n] * EyT[h][n]   (K-contiguous pairs)
          a[2 * jj]     = (bf16_t)((float)EyT[(h0 + lcol) * NPTS + k]     * u[k]);
          a[2 * jj + 1] = (bf16_t)((float)EyT[(h0 + lcol) * NPTS + k + 1] * u[k + 1]);
          // B[k=n][ncol=w] = Ex[n][w]
          const int kB = kb + ((jj < 4) ? 0 : 16) + hi * 8 + 2 * (jj & 3);
          b[2 * jj]     = Ex[kB * WW + (w0 + lcol)];
          b[2 * jj + 1] = Ex[(kB + 1) * WW + (w0 + lcol)];
        }
        c = __builtin_amdgcn_wmma_f32_16x16x32_bf16(false, a, false, b,
                                                    (short)0, c, false, false);
      }
#pragma unroll
      for (int i = 0; i < 8; ++i) {            // D: row = i + hi*8, col = lcol
        const int h = h0 + i + hi * 8;
        const int w = w0 + lcol;
        const float s  = c[i];
        const float bv = nd[h * WW + w];
        const float vf = bv / (s + MEPS);
        Vb[h * WW + w] = (bf16_t)vf;
        if (last) {
          float beta = __logf(vf + MEPS);
          beta = fminf(fmaxf(beta, -10000.0f), 10000.0f);
          ot_loc += bv * beta;
          sb_loc += ud[h * WW + w] * beta;
        }
      }
    }
    if (last) { atomicAdd(&scal[1], ot_loc); atomicAdd(&scal[2], sb_loc); }
    __syncthreads();
    if (tid < NPTS) r[tid] = 0.0f;
    __syncthreads();

    // ===== phase 2: T = Ex @ V^T ; r[n] = sum_h Ey[n,h]*T[n,h] =====
    {
      const int n0 = wave * 16;                // 32 n-tiles / 32 waves
      v16bf afr[4];
      v16bf a2fr[4] = {};
#pragma unroll
      for (int ks = 0; ks < 4; ++ks) {
        const int kb = ks * 32;
#pragma unroll
        for (int jj = 0; jj < 8; ++jj) {
          const int kloc = ((jj < 4) ? 0 : 16) + hi * 8 + 2 * (jj & 3);
          const int k = kb + kloc;             // k = w index
          const bf16_t e0 = Ex[(n0 + lcol) * WW + k];
          const bf16_t e1 = Ex[(n0 + lcol) * WW + k + 1];
          afr[ks][2 * jj] = e0;  afr[ks][2 * jj + 1] = e1;
          if (last) {                          // A2 = Ex .* xdis  (for wd)
            const float d0 = px[n0 + lcol] - xcoord(k);
            const float d1 = px[n0 + lcol] - xcoord(k + 1);
            a2fr[ks][2 * jj]     = (bf16_t)((float)e0 * d0 * d0);
            a2fr[ks][2 * jj + 1] = (bf16_t)((float)e1 * d1 * d1);
          }
        }
      }
      float racc[8], qacc[8];
#pragma unroll
      for (int i = 0; i < 8; ++i) { racc[i] = 0.0f; qacc[i] = 0.0f; }
      for (int ht = 0; ht < 8; ++ht) {
        const int h0 = ht * 16;
        v8f c = {}, c2 = {};
#pragma unroll
        for (int ks = 0; ks < 4; ++ks) {
          const int kb = ks * 32;
          v16bf b;                             // B[k=w][ncol=h] = Vb[h][w]
#pragma unroll
          for (int jj = 0; jj < 8; ++jj) {
            const int k = kb + ((jj < 4) ? 0 : 16) + hi * 8 + 2 * (jj & 3);
            b[2 * jj]     = Vb[(h0 + lcol) * WW + k];
            b[2 * jj + 1] = Vb[(h0 + lcol) * WW + k + 1];
          }
          c = __builtin_amdgcn_wmma_f32_16x16x32_bf16(false, afr[ks], false, b,
                                                      (short)0, c, false, false);
          if (last)
            c2 = __builtin_amdgcn_wmma_f32_16x16x32_bf16(false, a2fr[ks], false, b,
                                                         (short)0, c2, false, false);
        }
#pragma unroll
        for (int i = 0; i < 8; ++i) {
          const int n = n0 + i + hi * 8;
          const int h = h0 + lcol;
          const float ey = (float)EyT[h * NPTS + n];
          racc[i] += ey * c[i];
          if (last) {
            const float yd = py[n] - ycoord(h);
            qacc[i] += ey * (yd * yd * c[i] + c2[i]);
          }
        }
      }
#pragma unroll
      for (int i = 0; i < 8; ++i) {
        const int n = n0 + i + hi * 8;
        atomicAdd(&r[n], racc[i]);
        if (last) atomicAdd(&Q[n], qacc[i]);
      }
    }
    __syncthreads();
    if (tid < NPTS) {
      const float unew = (1.0f / NPTS) / (r[tid] + MEPS);
      u[tid] = unew;
      if (last) atomicAdd(&scal[3], unew * Q[tid]);   // wd = sum_n u[n]*Q[n]
    }
    __syncthreads();
  }

  if (tid == 0) {
    const float sc = scal[0], ot = scal[1], sb = scal[2], wd = scal[3];
    const float denom = sc * sc + 1e-8f;
    const float c1 = sc / denom, c2 = sb / denom;
    ws[img * 4 + 0] = c1 * sb - c2 * sc;   // loss_term (≈0, computed faithfully)
    ws[img * 4 + 1] = wd;
    ws[img * 4 + 2] = ot;
  }
}

__global__ void ot_finalize_kernel(const float* __restrict__ ws,
                                   float* __restrict__ out) {
  if (threadIdx.x == 0 && blockIdx.x == 0) {
    float loss = 0.0f, wd = 0.0f, ot = 0.0f;
    for (int b = 0; b < 4; ++b) {
      loss += ws[b * 4 + 0];
      wd   += ws[b * 4 + 1];
      ot   += ws[b * 4 + 2];
    }
    out[0] = loss * 0.25f;
    out[1] = wd;
    out[2] = ot * 0.25f;
    out[3] = (float)(NITER + 1);
  }
}

extern "C" void kernel_launch(void* const* d_in, const int* in_sizes, int n_in,
                              void* d_out, int out_size, void* d_ws, size_t ws_size,
                              hipStream_t stream) {
  (void)in_sizes; (void)n_in; (void)out_size; (void)ws_size;
  const float* nd  = (const float*)d_in[0];
  const float* ud  = (const float*)d_in[1];
  const float* pts = (const float*)d_in[2];
  float* ws = (float*)d_ws;

  ot_sinkhorn_kernel<<<dim3(4), dim3(TPB), SMEM_BYTES, stream>>>(nd, ud, pts, ws);
  ot_finalize_kernel<<<dim3(1), dim3(32), 0, stream>>>(ws, (float*)d_out);
}
